// StokenAttention_47665547051354
// MI455X (gfx1250) — compile-verified, matrix operand order checked by reading, not run on hardware
//
#include <hip/hip_runtime.h>
#include <hip/hip_bf16.h>

#define B_  2
#define N_  32768
#define C_  256
#define STK 16
#define M_  (N_ / STK)   // 2048
#define H_  8
#define HD  32

typedef __bf16 bf16;
typedef __attribute__((ext_vector_type(16))) __bf16 v16bf;
typedef __attribute__((ext_vector_type(8)))  float  v8f;

// ---------------------------------------------------------------------------
// Fragment loader for 16-bit WMMA operands (ISA 7.12.2 layout).
// A 16x32: lane L(0..15) row M=L, K={0..7,16..23}; lanes 16..31 same rows,
// K={8..15,24..31}. B 32x16 stored (n-line, k-contiguous) uses the same
// pattern with M<->N roles swapped. K pairs -> one dword load.
// ---------------------------------------------------------------------------
__device__ __forceinline__ v16bf load_frag16(const bf16* base, int ld, int row, int kk) {
  int lane = threadIdx.x & 31;
  int lrow = row + (lane & 15);
  int hi   = lane >> 4;
  union { v16bf v; unsigned u[8]; } f;
  const unsigned* p =
      reinterpret_cast<const unsigned*>(base + (size_t)lrow * ld + kk);
#pragma unroll
  for (int r = 0; r < 8; ++r) {
    int k = ((r < 4) ? (2 * r) : (8 + 2 * r)) + (hi ? 8 : 0);
    f.u[r] = p[k >> 1];
  }
  return f.v;
}

__device__ __forceinline__ v8f wmma_bf16(v16bf a, v16bf b, v8f c) {
  return __builtin_amdgcn_wmma_f32_16x16x32_bf16(false, a, false, b,
                                                 (short)0, c, false, false);
}

// ---------------------------------------------------------------------------
// Async global->LDS copy (16B per lane), ASYNCcnt-tracked (ISA 10. async ops)
// Generic LDS pointer low 32 bits == LDS byte offset (aperture mapping).
// ---------------------------------------------------------------------------
__device__ __forceinline__ void async_copy16(const void* gsrc, void* ldst) {
  unsigned lds_off = (unsigned)(unsigned long long)ldst;
  unsigned long long ga = (unsigned long long)gsrc;
  asm volatile("global_load_async_to_lds_b128 %0, %1, off"
               :: "v"(lds_off), "v"(ga) : "memory");
}
__device__ __forceinline__ void wait_async0() {
#if __has_builtin(__builtin_amdgcn_s_wait_asynccnt)
  __builtin_amdgcn_s_wait_asynccnt(0);
#else
  asm volatile("s_wait_asynccnt 0" ::: "memory");
#endif
}

// ---------------------------------------------------------------------------
// 0) Convert weights to bf16
// ---------------------------------------------------------------------------
__global__ void k_convert(const float* __restrict__ Wqkv,
                          const float* __restrict__ Wproj,
                          bf16* __restrict__ Wq_b, bf16* __restrict__ Wp_b) {
  int i = blockIdx.x * blockDim.x + threadIdx.x;
  if (i < 768 * 256) Wq_b[i] = (bf16)Wqkv[i];
  if (i < 256 * 256) Wp_b[i] = (bf16)Wproj[i];
}

// ---------------------------------------------------------------------------
// 1) sf[b, mm, c] = mean over 16 pixels of window   (B,m,C layout)
// ---------------------------------------------------------------------------
__global__ void k_sf(const float* __restrict__ x, float* __restrict__ sf) {
  int b = blockIdx.x / M_, mm = blockIdx.x % M_;
  int c = threadIdx.x;
  const float* xp = x + ((size_t)b * N_ + (size_t)mm * STK) * C_ + c;
  float s = 0.f;
#pragma unroll
  for (int j = 0; j < STK; ++j) s += xp[(size_t)j * C_];
  sf[((size_t)b * M_ + mm) * C_ + c] = s * (1.f / 16.f);
}

// ---------------------------------------------------------------------------
// 2) aff[b,mm,j,k] = softmax_k( scale * <pix_j, sf[mm+k-1]> ), winsum = sum_j
//    x window tile (16x256 f32, contiguous 16KB) staged via async->LDS.
// ---------------------------------------------------------------------------
__global__ void k_aff(const float* __restrict__ x, const float* __restrict__ sf,
                      float* __restrict__ aff, float* __restrict__ winsum) {
  int b = blockIdx.x / M_, mm = blockIdx.x % M_;
  int tid = threadIdx.x;
  int j = tid >> 4, cs = tid & 15;
  __shared__ float xs[16][256];
  __shared__ float part[16][16][3];
  __shared__ float logit[16][3];
  __shared__ float av[16][3];

  const char* gsrc =
      (const char*)(x + ((size_t)b * N_ + (size_t)mm * STK) * C_);
  char* ldst = (char*)&xs[0][0];
#pragma unroll
  for (int q = 0; q < 4; ++q) {
    int i = tid + q * 256;
    async_copy16(gsrc + (size_t)i * 16, ldst + (size_t)i * 16);
  }
  wait_async0();
  __syncthreads();

  float xv[16];
#pragma unroll
  for (int i = 0; i < 16; ++i) xv[i] = xs[j][cs * 16 + i];

  float p0 = 0.f, p1 = 0.f, p2 = 0.f;
#pragma unroll
  for (int k = 0; k < 3; ++k) {
    int w = mm + k - 1;
    if (w >= 0 && w < M_) {
      const float* sr = sf + ((size_t)b * M_ + w) * C_ + cs * 16;
      float s = 0.f;
#pragma unroll
      for (int i = 0; i < 16; ++i) s += xv[i] * sr[i];
      if (k == 0) p0 = s; else if (k == 1) p1 = s; else p2 = s;
    }
  }
  part[j][cs][0] = p0; part[j][cs][1] = p1; part[j][cs][2] = p2;
  __syncthreads();
  if (cs < 3) {
    float s = 0.f;
#pragma unroll
    for (int i = 0; i < 16; ++i) s += part[j][i][cs];
    logit[j][cs] = s * 0.0625f;  // C^-0.5, C=256
  }
  __syncthreads();
  if (cs == 0) {
    float l0 = logit[j][0], l1 = logit[j][1], l2 = logit[j][2];
    float mx = fmaxf(l0, fmaxf(l1, l2));
    float e0 = __expf(l0 - mx), e1 = __expf(l1 - mx), e2 = __expf(l2 - mx);
    float inv = 1.f / (e0 + e1 + e2);
    float a0 = e0 * inv, a1 = e1 * inv, a2 = e2 * inv;
    size_t base = (((size_t)b * M_ + mm) * 16 + j) * 3;
    aff[base] = a0; aff[base + 1] = a1; aff[base + 2] = a2;
    av[j][0] = a0; av[j][1] = a1; av[j][2] = a2;
  }
  __syncthreads();
  if (tid < 3) {
    float s = 0.f;
#pragma unroll
    for (int i = 0; i < 16; ++i) s += av[i][tid];
    winsum[((size_t)b * M_ + mm) * 3 + tid] = s;
  }
}

// ---------------------------------------------------------------------------
// 3) sf2 (folded, normalized) -> bf16 (B,m,C layout)
// ---------------------------------------------------------------------------
__global__ void k_sf2(const float* __restrict__ x, const float* __restrict__ aff,
                      const float* __restrict__ winsum, bf16* __restrict__ sf2b) {
  int b = blockIdx.x / M_, mm = blockIdx.x % M_;
  int c = threadIdx.x;
  __shared__ float wa[3][16];
  __shared__ float denom_s;
  if (threadIdx.x < 48) {
    int w = threadIdx.x / 16, j = threadIdx.x % 16;  // k == w, src = mm+1-w
    int src = mm + 1 - w;
    float v = 0.f;
    if (src >= 0 && src < M_)
      v = aff[(((size_t)b * M_ + src) * 16 + j) * 3 + w];
    wa[w][j] = v;
  }
  if (threadIdx.x == 255) {
    float d = winsum[((size_t)b * M_ + mm) * 3 + 1];
    if (mm + 1 < M_) d += winsum[((size_t)b * M_ + mm + 1) * 3 + 0];
    if (mm > 0)      d += winsum[((size_t)b * M_ + mm - 1) * 3 + 2];
    denom_s = d + 1e-12f;
  }
  __syncthreads();
  float acc = 0.f;
#pragma unroll
  for (int w = 0; w < 3; ++w) {
    int src = mm + 1 - w;
    if (src >= 0 && src < M_) {
      const float* xr = x + ((size_t)b * N_ + (size_t)src * STK) * C_ + c;
#pragma unroll
      for (int j = 0; j < 16; ++j) acc += wa[w][j] * xr[(size_t)j * C_];
    }
  }
  sf2b[((size_t)b * M_ + mm) * C_ + c] = (bf16)(acc / denom_s);
}

// ---------------------------------------------------------------------------
// 4) QKV GEMM, 16x64 tile per wave (A fragment reused 4x).
// ---------------------------------------------------------------------------
__global__ void k_qkv(const bf16* __restrict__ Wq, const bf16* __restrict__ sf2b,
                      bf16* __restrict__ qT, bf16* __restrict__ kT,
                      bf16* __restrict__ vL) {
  int id = blockIdx.x;                  // B * 48 * 32
  int b = id / (48 * 32);
  int rem = id % (48 * 32);
  int ot = rem / 32, nt4 = rem % 32;
  const bf16* Bb = sf2b + (size_t)b * M_ * C_;
  v8f acc[4] = {{}, {}, {}, {}};
  for (int kk = 0; kk < C_; kk += 32) {
    v16bf a = load_frag16(Wq, C_, ot * 16, kk);
#pragma unroll
    for (int t = 0; t < 4; ++t) {
      v16bf bb = load_frag16(Bb, C_, nt4 * 64 + t * 16, kk);
      acc[t] = wmma_bf16(a, bb, acc[t]);
    }
  }
  int lane = threadIdx.x & 31, hi = lane >> 4;
#pragma unroll
  for (int t = 0; t < 4; ++t) {
    int col = nt4 * 64 + t * 16 + (lane & 15);
#pragma unroll
    for (int r = 0; r < 8; ++r) {
      int o = ot * 16 + r + hi * 8;
      int h = o / 96, w = o % 96;
      bf16 v = (bf16)acc[t][r];
      if (w < 32)
        qT[(((size_t)b * H_ + h) * M_ + col) * HD + w] = v;
      else if (w < 64)
        kT[(((size_t)b * H_ + h) * M_ + col) * HD + (w - 32)] = v;
      else
        vL[(((size_t)b * H_ + h) * HD + (w - 64)) * M_ + col] = v;
    }
  }
}

// ---------------------------------------------------------------------------
// 5) Row stats of S = scale*K^T Q: 32 key rows per wave, Q fragment reused 2x.
// ---------------------------------------------------------------------------
__global__ void k_rowstats(const bf16* __restrict__ qT, const bf16* __restrict__ kT,
                           float* __restrict__ rowmax, float* __restrict__ rowsum) {
  int id = blockIdx.x;                  // B * H * (M/32)
  int b = id / (H_ * (M_ / 32));
  int rem = id % (H_ * (M_ / 32));
  int h = rem / (M_ / 32), mt2 = rem % (M_ / 32);
  size_t base = ((size_t)b * H_ + h) * M_ * HD;
  const float scale = 0.1767766952966369f;  // hd^-0.5

  v16bf a0 = load_frag16(kT + base, HD, mt2 * 32, 0);
  v16bf a1 = load_frag16(kT + base, HD, mt2 * 32 + 16, 0);
  float lm[2][8], ls[2][8];
#pragma unroll
  for (int t = 0; t < 2; ++t)
#pragma unroll
    for (int r = 0; r < 8; ++r) { lm[t][r] = -1e30f; ls[t][r] = 0.f; }

  for (int nt = 0; nt < M_ / 16; ++nt) {
    v16bf bb = load_frag16(qT + base, HD, nt * 16, 0);
    v8f z = {};
    v8f s0 = wmma_bf16(a0, bb, z);
    v8f s1 = wmma_bf16(a1, bb, z);
#pragma unroll
    for (int r = 0; r < 8; ++r) {
      float x0 = s0[r] * scale;
      float n0 = fmaxf(lm[0][r], x0);
      ls[0][r] = ls[0][r] * __expf(lm[0][r] - n0) + __expf(x0 - n0);
      lm[0][r] = n0;
      float x1 = s1[r] * scale;
      float n1 = fmaxf(lm[1][r], x1);
      ls[1][r] = ls[1][r] * __expf(lm[1][r] - n1) + __expf(x1 - n1);
      lm[1][r] = n1;
    }
  }
#pragma unroll
  for (int t = 0; t < 2; ++t)
#pragma unroll
    for (int r = 0; r < 8; ++r) {
#pragma unroll
      for (int msk = 1; msk < 16; msk <<= 1) {
        float om = __shfl_xor(lm[t][r], msk, 32);
        float os = __shfl_xor(ls[t][r], msk, 32);
        float nm = fmaxf(lm[t][r], om);
        ls[t][r] = ls[t][r] * __expf(lm[t][r] - nm) + os * __expf(om - nm);
        lm[t][r] = nm;
      }
    }
  int lane = threadIdx.x & 31;
  if ((lane & 15) == 0) {
    int hi = lane >> 4;
#pragma unroll
    for (int t = 0; t < 2; ++t) {
      size_t rb = ((size_t)b * H_ + h) * M_ + mt2 * 32 + t * 16 + hi * 8;
#pragma unroll
      for (int r = 0; r < 8; ++r) { rowmax[rb + r] = lm[t][r]; rowsum[rb + r] = ls[t][r]; }
    }
  }
}

// ---------------------------------------------------------------------------
// 6) out[d,n] = sum_m v[d,m] P[m,n]; 32 queries per wave, stats in LDS,
//    P staged through LDS in B-fragment order. 8 WMMA per 32-key step.
// ---------------------------------------------------------------------------
__global__ void k_attn_out(const bf16* __restrict__ qT, const bf16* __restrict__ kT,
                           const bf16* __restrict__ vL,
                           const float* __restrict__ rowmax,
                           const float* __restrict__ rowsum,
                           bf16* __restrict__ attnT) {
  int id = blockIdx.x;                  // B * H * (M/32)
  int b = id / (H_ * (M_ / 32));
  int rem = id % (H_ * (M_ / 32));
  int h = rem / (M_ / 32), nt2 = rem % (M_ / 32);
  size_t base = ((size_t)b * H_ + h) * M_ * HD;
  size_t rbase = ((size_t)b * H_ + h) * M_;
  const bf16* vbase = vL + ((size_t)b * H_ + h) * HD * M_;
  const float scale = 0.1767766952966369f;

  __shared__ float ldsRM[M_];
  __shared__ float ldsRI[M_];
  __shared__ bf16 Plds0[16 * 34];
  __shared__ bf16 Plds1[16 * 34];

  for (int i = threadIdx.x; i < M_; i += 32) {
    ldsRM[i] = rowmax[rbase + i];
    ldsRI[i] = 1.f / rowsum[rbase + i];
  }
  __syncthreads();

  v16bf bq0 = load_frag16(qT + base, HD, nt2 * 32, 0);
  v16bf bq1 = load_frag16(qT + base, HD, nt2 * 32 + 16, 0);
  v8f acc[2][2] = {{{}, {}}, {{}, {}}};
  int lane = threadIdx.x & 31, hi = lane >> 4, colL = lane & 15;

  for (int mi0 = 0; mi0 < M_; mi0 += 32) {
#pragma unroll
    for (int t = 0; t < 2; ++t) {
      v16bf ak = load_frag16(kT + base, HD, mi0 + t * 16, 0);
      v8f z = {};
      v8f s0 = wmma_bf16(ak, bq0, z);
      v8f s1 = wmma_bf16(ak, bq1, z);
#pragma unroll
      for (int r = 0; r < 8; ++r) {
        int row = mi0 + t * 16 + r + hi * 8;
        float rm = ldsRM[row], ri = ldsRI[row];
        int pidx = colL * 34 + t * 16 + r + hi * 8;
        Plds0[pidx] = (bf16)(__expf(s0[r] * scale - rm) * ri);
        Plds1[pidx] = (bf16)(__expf(s1[r] * scale - rm) * ri);
      }
    }
    __syncthreads();
    v16bf pf0 = load_frag16(Plds0, 34, 0, 0);
    v16bf pf1 = load_frag16(Plds1, 34, 0, 0);
    v16bf va0 = load_frag16(vbase, M_, 0, mi0);
    v16bf va1 = load_frag16(vbase, M_, 16, mi0);
    acc[0][0] = wmma_bf16(va0, pf0, acc[0][0]);
    acc[0][1] = wmma_bf16(va1, pf0, acc[0][1]);
    acc[1][0] = wmma_bf16(va0, pf1, acc[1][0]);
    acc[1][1] = wmma_bf16(va1, pf1, acc[1][1]);
    __syncthreads();
  }
#pragma unroll
  for (int tq = 0; tq < 2; ++tq) {
    int col = nt2 * 32 + tq * 16 + colL;
#pragma unroll
    for (int r = 0; r < 8; ++r) {
      attnT[((size_t)b * M_ + col) * C_ + h * HD + r + hi * 8] = (bf16)acc[tq][0][r];
      attnT[((size_t)b * M_ + col) * C_ + h * HD + 16 + r + hi * 8] = (bf16)acc[tq][1][r];
    }
  }
}

// ---------------------------------------------------------------------------
// 7) Projection GEMM + bias -> sf3 (B,m,C) f32; 16x64 tile per wave.
// ---------------------------------------------------------------------------
__global__ void k_proj(const bf16* __restrict__ Wp, const bf16* __restrict__ attnT,
                       const float* __restrict__ bias, float* __restrict__ sf3) {
  int id = blockIdx.x;                  // B * 16 * 32
  int b = id / (16 * 32);
  int rem = id % (16 * 32);
  int ct = rem / 32, nt4 = rem % 32;
  const bf16* Bb = attnT + (size_t)b * M_ * C_;
  v8f acc[4] = {{}, {}, {}, {}};
  for (int kk = 0; kk < C_; kk += 32) {
    v16bf a = load_frag16(Wp, C_, ct * 16, kk);
#pragma unroll
    for (int t = 0; t < 4; ++t) {
      v16bf bb = load_frag16(Bb, C_, nt4 * 64 + t * 16, kk);
      acc[t] = wmma_bf16(a, bb, acc[t]);
    }
  }
  int lane = threadIdx.x & 31, hi = lane >> 4;
#pragma unroll
  for (int t = 0; t < 4; ++t) {
    int col = nt4 * 64 + t * 16 + (lane & 15);
#pragma unroll
    for (int r = 0; r < 8; ++r) {
      int c = ct * 16 + r + hi * 8;
      sf3[((size_t)b * M_ + col) * C_ + c] = acc[t][r] + bias[c];
    }
  }
}

// ---------------------------------------------------------------------------
// 8) Scatter: out[b,c,mm*16+j] = sum_k aff[b,mm,j,k] * sf3[b, mm+k-1, c]
// ---------------------------------------------------------------------------
__global__ void k_final(const float* __restrict__ aff, const float* __restrict__ sf3,
                        float* __restrict__ out) {
  int b = blockIdx.x / M_, mm = blockIdx.x % M_;
  __shared__ float wa[3][16];
  if (threadIdx.x < 48) {
    int j = threadIdx.x / 3, k = threadIdx.x % 3;
    wa[k][j] = aff[(((size_t)b * M_ + mm) * 16 + j) * 3 + k];
  }
  __syncthreads();
  int j = threadIdx.x & 15, cc = threadIdx.x >> 4;
  const float* s1 = sf3 + ((size_t)b * M_ + mm) * C_;
  const float* s0 = (mm > 0) ? s1 - C_ : nullptr;
  const float* s2 = (mm + 1 < M_) ? s1 + C_ : nullptr;
  float a0 = wa[0][j], a1 = wa[1][j], a2 = wa[2][j];
#pragma unroll
  for (int ci = 0; ci < 16; ++ci) {
    int c = cc + ci * 16;
    float v = a1 * s1[c];
    if (s0) v += a0 * s0[c];
    if (s2) v += a2 * s2[c];
    out[(size_t)b * C_ * N_ + (size_t)c * N_ + (size_t)mm * STK + j] = v;
  }
}

// ---------------------------------------------------------------------------
extern "C" void kernel_launch(void* const* d_in, const int* in_sizes, int n_in,
                              void* d_out, int out_size, void* d_ws, size_t ws_size,
                              hipStream_t stream) {
  (void)in_sizes; (void)n_in; (void)out_size; (void)ws_size;
  const float* x     = (const float*)d_in[0];
  const float* Wqkv  = (const float*)d_in[1];
  const float* Wproj = (const float*)d_in[2];
  const float* bproj = (const float*)d_in[3];
  float* out = (float*)d_out;

  char* w = (char*)d_ws;
  auto alloc = [&](size_t bytes) -> void* {
    void* p = (void*)w;
    w += (bytes + 255) & ~(size_t)255;
    return p;
  };
  float* sf     = (float*)alloc((size_t)B_ * M_ * C_ * 4);
  float* aff    = (float*)alloc((size_t)B_ * M_ * 16 * 3 * 4);
  float* winsum = (float*)alloc((size_t)B_ * M_ * 3 * 4);
  float* rowmax = (float*)alloc((size_t)B_ * H_ * M_ * 4);
  float* rowsum = (float*)alloc((size_t)B_ * H_ * M_ * 4);
  float* sf3    = (float*)alloc((size_t)B_ * M_ * C_ * 4);
  bf16*  sf2b   = (bf16*)alloc((size_t)B_ * M_ * C_ * 2);
  bf16*  Wq_b   = (bf16*)alloc((size_t)768 * 256 * 2);
  bf16*  Wp_b   = (bf16*)alloc((size_t)256 * 256 * 2);
  bf16*  qT     = (bf16*)alloc((size_t)B_ * H_ * M_ * HD * 2);
  bf16*  kT     = (bf16*)alloc((size_t)B_ * H_ * M_ * HD * 2);
  bf16*  vL     = (bf16*)alloc((size_t)B_ * H_ * M_ * HD * 2);
  bf16*  attnT  = (bf16*)alloc((size_t)B_ * M_ * C_ * 2);

  hipLaunchKernelGGL(k_convert, dim3(768), dim3(256), 0, stream, Wqkv, Wproj, Wq_b, Wp_b);
  hipLaunchKernelGGL(k_sf,      dim3(B_ * M_), dim3(256), 0, stream, x, sf);
  hipLaunchKernelGGL(k_aff,     dim3(B_ * M_), dim3(256), 0, stream, x, sf, aff, winsum);
  hipLaunchKernelGGL(k_sf2,     dim3(B_ * M_), dim3(256), 0, stream, x, aff, winsum, sf2b);
  hipLaunchKernelGGL(k_qkv,     dim3(B_ * 48 * 32), dim3(32), 0, stream, Wq_b, sf2b, qT, kT, vL);
  hipLaunchKernelGGL(k_rowstats, dim3(B_ * H_ * (M_ / 32)), dim3(32), 0, stream, qT, kT, rowmax, rowsum);
  hipLaunchKernelGGL(k_attn_out, dim3(B_ * H_ * (M_ / 32)), dim3(32), 0, stream, qT, kT, vL, rowmax, rowsum, attnT);
  hipLaunchKernelGGL(k_proj,    dim3(B_ * 16 * 32), dim3(32), 0, stream, Wp_b, attnT, bproj, sf3);
  hipLaunchKernelGGL(k_final,   dim3(B_ * M_), dim3(256), 0, stream, aff, sf3, out);
}
